// MPLNNRegressor_18107582119955
// MI455X (gfx1250) — compile-verified, matrix operand order checked by reading, not run on hardware
//
#include <hip/hip_runtime.h>
#include <hip/hip_bf16.h>
#include <stdint.h>

#define NNODES 30000
#define NEDGES 480000
#define NGRAPHS 64

typedef __attribute__((ext_vector_type(16))) __bf16 v16bf;
typedef __attribute__((ext_vector_type(8)))  float  v8f;

// ---------------- workspace layout (bytes) ----------------
static constexpr size_t SB        = (size_t)NNODES * 128 * 4;   // 15,360,000
static constexpr size_t OFF_BIG0  = 0;
static constexpr size_t OFF_BIG1  = SB;
static constexpr size_t OFF_BIG2  = 2 * SB;
static constexpr size_t OFF_EW    = 3 * SB;                      // E floats
static constexpr size_t OFF_SEGMX = OFF_EW    + (size_t)NEDGES * 4;
static constexpr size_t OFF_DENOM = OFF_SEGMX + (size_t)NNODES * 4;
static constexpr size_t OFF_W1T   = OFF_DENOM + (size_t)NNODES * 4;  // 16-aligned
static constexpr size_t OFF_W2T   = OFF_W1T + 32768;
static constexpr size_t OFF_W5T   = OFF_W2T + 16384;
static constexpr size_t OFF_W6T   = OFF_W5T + 4096;
static constexpr size_t OFF_X1    = OFF_W6T + 4096;
static constexpr size_t OFF_X2    = OFF_X1  + 16384;

// ---------------- device helpers ----------------
__device__ __forceinline__ v8f wmma_bf16(v16bf a, v16bf b, v8f c) {
  return __builtin_amdgcn_wmma_f32_16x16x32_bf16(false, a, false, b,
                                                 (short)0, c, false, false);
}

// order-preserving float<->uint encoding for atomicMax-based segment max
__device__ __forceinline__ unsigned fenc(float f) {
  unsigned u = __float_as_uint(f);
  return (u & 0x80000000u) ? ~u : (u | 0x80000000u);
}
__device__ __forceinline__ float fdec(unsigned e) {
  unsigned u = (e & 0x80000000u) ? (e ^ 0x80000000u) : ~e;
  return __uint_as_float(u);
}

// A fragment (16x32 bf16 tile) for edge rows: row = [x_i | x_j - x_i]
template<int CI>
__device__ __forceinline__ v16bf fragA_edge(const float* xi, const float* xj,
                                            int kc, int hi) {
  v16bf a;
  const int kb0 = kc * 32 + hi * 8;
  const int kb1 = kb0 + 16;
#pragma unroll
  for (int j = 0; j < 8; ++j) {
    int k = kb0 + j;
    float v = (k < CI) ? xi[k] : (xj[k - CI] - xi[k - CI]);
    a[j] = (__bf16)v;
  }
#pragma unroll
  for (int j = 0; j < 8; ++j) {
    int k = kb1 + j;
    float v = (k < CI) ? xi[k] : (xj[k - CI] - xi[k - CI]);
    a[j + 8] = (__bf16)v;
  }
  return a;
}

// A fragment from a contiguous f32 row (converted to bf16 in-register)
__device__ __forceinline__ v16bf fragA_f32row(const float* row, int kc, int hi) {
  v16bf a;
  const int kb0 = kc * 32 + hi * 8;
  const int kb1 = kb0 + 16;
#pragma unroll
  for (int j = 0; j < 8; ++j) a[j]     = (__bf16)row[kb0 + j];
#pragma unroll
  for (int j = 0; j < 8; ++j) a[j + 8] = (__bf16)row[kb1 + j];
  return a;
}

// A/B fragment from a contiguous bf16 row (16B-aligned): two b128 loads
__device__ __forceinline__ v16bf fragRow_bf16(const __bf16* row, int kc, int hi) {
  union { v16bf v; uint4 q[2]; } u;
  const __bf16* r = row + kc * 32;
  u.q[0] = *(const uint4*)(r + hi * 8);
  u.q[1] = *(const uint4*)(r + 16 + hi * 8);
  return u.v;
}
__device__ __forceinline__ v16bf fragB(const __bf16* Wt, int ldK, int n,
                                       int kc, int hi) {
  return fragRow_bf16(Wt + (size_t)n * ldK, kc, hi);
}

// ---------------- small utility kernels ----------------
__global__ void k_transpose_bf16(__bf16* __restrict__ dst,
                                 const float* __restrict__ src, int R, int C) {
  int i = blockIdx.x * blockDim.x + threadIdx.x;
  if (i >= R * C) return;
  int r = i / C, c = i % C;
  dst[(size_t)c * R + r] = (__bf16)src[i];
}

__global__ void k_edge_exp(float* __restrict__ ew,
                           const unsigned* __restrict__ segmax,
                           float* __restrict__ denom,
                           const int* __restrict__ src) {
  int e = blockIdx.x * blockDim.x + threadIdx.x;
  if (e >= NEDGES) return;
  int s = src[e];
  float v = expf(ew[e] - fdec(segmax[s]));
  ew[e] = v;
  atomicAdd(&denom[s], v);
}

__global__ void k_edge_div(float* __restrict__ ew,
                           const float* __restrict__ denom,
                           const int* __restrict__ src) {
  int e = blockIdx.x * blockDim.x + threadIdx.x;
  if (e >= NEDGES) return;
  ew[e] = ew[e] / denom[src[e]];
}

// ---------------- edge pass 1: msg GEMM + attention logits ----------------
template<int CI, int CO>
__global__ void __launch_bounds__(128)
k_edge_pass1(const float* __restrict__ x,
             const int* __restrict__ src, const int* __restrict__ dst,
             const __bf16* __restrict__ w1t, const float* __restrict__ b1,
             const __bf16* __restrict__ w5t, const float* __restrict__ b5,
             const __bf16* __restrict__ w6t, const float* __restrict__ b6,
             const float* __restrict__ w7, const float* __restrict__ b7,
             float* __restrict__ wlogit, unsigned* __restrict__ segmax,
             int nTiles) {
  constexpr int K1  = 2 * CI;
  constexpr int KC1 = K1 / 32;
  constexpr int NT  = CO / 16;
  constexpr int KC6 = CO / 32;
  constexpr int KC5 = CI / 32;
  __shared__ alignas(16) __bf16 smsg[4][16 * CO];

  const int wave = threadIdx.x >> 5;
  const int lane = threadIdx.x & 31;
  const int tile = blockIdx.x * 4 + wave;
  if (tile >= nTiles) return;                 // wave-uniform exit
  const int base = tile * 16;
  const int m = lane & 15;
  const int hi = lane >> 4;
  const int e0 = base + m;
  const int ni = dst[e0];                      // x_i = x[dst]
  const int nj = src[e0];                      // x_j = x[src]
  const float* xi = x + (size_t)ni * CI;
  const float* xj = x + (size_t)nj * CI;

  v16bf a1[KC1];
#pragma unroll
  for (int kc = 0; kc < KC1; ++kc) a1[kc] = fragA_edge<CI>(xi, xj, kc, hi);

  __bf16* mbase = &smsg[wave][0];
#pragma unroll
  for (int nt = 0; nt < NT; ++nt) {
    v8f acc = {};
#pragma unroll
    for (int kc = 0; kc < KC1; ++kc)
      acc = wmma_bf16(a1[kc], fragB(w1t, K1, nt * 16 + m, kc, hi), acc);
    const int n = nt * 16 + m;
    const float bias = b1[n];
#pragma unroll
    for (int r = 0; r < 8; ++r) {
      float v = acc[r] + bias;
      v = v > 0.f ? v : 0.f;                   // relu -> msg
      mbase[(r + 8 * hi) * CO + n] = (__bf16)v;
    }
  }

  // w2 = tanh(msg @ W6 + b6)   (A from LDS msg tile)
  v8f acc2 = {};
  const __bf16* mrow = mbase + m * CO;
#pragma unroll
  for (int kc = 0; kc < KC6; ++kc)
    acc2 = wmma_bf16(fragRow_bf16(mrow, kc, hi), fragB(w6t, CO, m, kc, hi), acc2);

  // w1 = tanh(x_i @ W5 + b5)
  v8f acc1 = {};
#pragma unroll
  for (int kc = 0; kc < KC5; ++kc)
    acc1 = wmma_bf16(fragA_f32row(xi, kc, hi), fragB(w5t, CI, m, kc, hi), acc1);

  const float b5n = b5[m], b6n = b6[m], w7n = w7[m];
  float p[8];
#pragma unroll
  for (int r = 0; r < 8; ++r)
    p[r] = tanhf(acc1[r] + b5n) * tanhf(acc2[r] + b6n) * w7n;
#pragma unroll
  for (int msk = 1; msk < 16; msk <<= 1) {
#pragma unroll
    for (int r = 0; r < 8; ++r) p[r] += __shfl_xor(p[r], msk, 16);
  }
  if (m == 0) {                                // lanes 0 and 16
    const float bb = b7[0];
#pragma unroll
    for (int r = 0; r < 8; ++r) {
      const int e = base + r + 8 * hi;
      const float wl = p[r] + bb;
      wlogit[e] = wl;
      atomicMax(&segmax[src[e]], fenc(wl));    // softmax grouped by SRC
    }
  }
}

// ---------------- edge pass 2: recompute msg, weighted scatter-add --------
template<int CI, int CO>
__global__ void __launch_bounds__(128)
k_edge_pass2(const float* __restrict__ x,
             const int* __restrict__ src, const int* __restrict__ dst,
             const __bf16* __restrict__ w1t, const float* __restrict__ b1,
             const float* __restrict__ ew, float* __restrict__ agg,
             int nTiles) {
  constexpr int K1  = 2 * CI;
  constexpr int KC1 = K1 / 32;
  constexpr int NT  = CO / 16;

  const int wave = threadIdx.x >> 5;
  const int lane = threadIdx.x & 31;
  const int tile = blockIdx.x * 4 + wave;
  if (tile >= nTiles) return;
  const int base = tile * 16;
  const int m = lane & 15;
  const int hi = lane >> 4;
  const int e0 = base + m;
  const float* xi = x + (size_t)dst[e0] * CI;
  const float* xj = x + (size_t)src[e0] * CI;

  v16bf a1[KC1];
#pragma unroll
  for (int kc = 0; kc < KC1; ++kc) a1[kc] = fragA_edge<CI>(xi, xj, kc, hi);

  float we[8]; int dn[8];
#pragma unroll
  for (int r = 0; r < 8; ++r) {
    const int e = base + r + 8 * hi;
    we[r] = ew[e];
    dn[r] = dst[e];
  }
#pragma unroll
  for (int nt = 0; nt < NT; ++nt) {
    v8f acc = {};
#pragma unroll
    for (int kc = 0; kc < KC1; ++kc)
      acc = wmma_bf16(a1[kc], fragB(w1t, K1, nt * 16 + m, kc, hi), acc);
    const int n = nt * 16 + m;
    const float bias = b1[n];
#pragma unroll
    for (int r = 0; r < 8; ++r) {
      float v = acc[r] + bias;
      v = v > 0.f ? v : 0.f;
      atomicAdd(&agg[(size_t)dn[r] * CO + n], v * we[r]);
    }
  }
}

// ---------------- per-node xn = relu(x @ W2 + b2) ----------------
template<int CI, int CO>
__global__ void __launch_bounds__(128)
k_node_xn(const float* __restrict__ x, const __bf16* __restrict__ w2t,
          const float* __restrict__ b2, float* __restrict__ xn, int nTiles) {
  constexpr int KC = CI / 32;
  constexpr int NT = CO / 16;
  const int wave = threadIdx.x >> 5;
  const int lane = threadIdx.x & 31;
  const int tile = blockIdx.x * 4 + wave;
  if (tile >= nTiles) return;
  const int base = tile * 16;
  const int m = lane & 15;
  const int hi = lane >> 4;
  const float* row = x + (size_t)(base + m) * CI;

  v16bf a[KC];
#pragma unroll
  for (int kc = 0; kc < KC; ++kc) a[kc] = fragA_f32row(row, kc, hi);
#pragma unroll
  for (int nt = 0; nt < NT; ++nt) {
    v8f acc = {};
#pragma unroll
    for (int kc = 0; kc < KC; ++kc)
      acc = wmma_bf16(a[kc], fragB(w2t, CI, nt * 16 + m, kc, hi), acc);
    const int n = nt * 16 + m;
    const float bias = b2[n];
#pragma unroll
    for (int r = 0; r < 8; ++r) {
      float v = acc[r] + bias;
      v = v > 0.f ? v : 0.f;
      xn[(size_t)(base + r + 8 * hi) * CO + n] = v;
    }
  }
}

// ---------------- gating combine: out = relu(a1*agg + a2*xn) (in-place agg)
template<int CO>
__global__ void k_combine(const float* __restrict__ xn, float* __restrict__ agg,
                          const float* __restrict__ w3, const float* __restrict__ b3,
                          const float* __restrict__ w4, const float* __restrict__ b4,
                          int nNodes) {
  constexpr int CPL = CO / 32;
  const int wave = threadIdx.x >> 5;
  const int lane = threadIdx.x & 31;
  const int n = blockIdx.x * (blockDim.x >> 5) + wave;
  if (n >= nNodes) return;
  float xv[CPL], av[CPL];
  float s3 = 0.f, s4 = 0.f;
#pragma unroll
  for (int j = 0; j < CPL; ++j) {
    const int c = lane + j * 32;
    xv[j] = xn[(size_t)n * CO + c];
    av[j] = agg[(size_t)n * CO + c];
    s3 += xv[j] * w3[c] + av[j] * w3[CO + c];  // cat = [xn, agg]
    s4 += xv[j] * w4[c] + av[j] * w4[CO + c];
  }
#pragma unroll
  for (int msk = 16; msk >= 1; msk >>= 1) {
    s3 += __shfl_xor(s3, msk, 32);
    s4 += __shfl_xor(s4, msk, 32);
  }
  const float a1 = 1.f / (1.f + expf(-(s3 + b3[0])));
  const float a2 = 1.f / (1.f + expf(-(s4 + b4[0])));
#pragma unroll
  for (int j = 0; j < CPL; ++j) {
    float o = a1 * av[j] + a2 * xv[j];
    o = o > 0.f ? o : 0.f;                     // outer relu of the layer
    agg[(size_t)n * CO + lane + j * 32] = o;
  }
}

// ---------------- attention pooling (64 channels) ----------------
__global__ void k_pool_g(const float* __restrict__ x, const float* __restrict__ gw,
                         const float* __restrict__ gb, const int* __restrict__ batch,
                         float* __restrict__ gbuf, unsigned* __restrict__ gmax) {
  const int wave = threadIdx.x >> 5;
  const int lane = threadIdx.x & 31;
  const int n = blockIdx.x * (blockDim.x >> 5) + wave;
  if (n >= NNODES) return;
  float s = x[(size_t)n * 64 + lane] * gw[lane] +
            x[(size_t)n * 64 + lane + 32] * gw[lane + 32];
#pragma unroll
  for (int msk = 16; msk >= 1; msk >>= 1) s += __shfl_xor(s, msk, 32);
  if (lane == 0) {
    const float g = s + gb[0];
    gbuf[n] = g;
    atomicMax(&gmax[batch[n]], fenc(g));
  }
}

__global__ void k_pool_e(float* __restrict__ gbuf, const unsigned* __restrict__ gmax,
                         float* __restrict__ gden, const int* __restrict__ batch) {
  int i = blockIdx.x * blockDim.x + threadIdx.x;
  if (i >= NNODES) return;
  int b = batch[i];
  float v = expf(gbuf[i] - fdec(gmax[b]));
  gbuf[i] = v;
  atomicAdd(&gden[b], v);
}

__global__ void k_pool_acc(const float* __restrict__ x, const float* __restrict__ gbuf,
                           const float* __restrict__ gden, const int* __restrict__ batch,
                           float* __restrict__ pool) {
  const int wave = threadIdx.x >> 5;
  const int lane = threadIdx.x & 31;
  const int n = blockIdx.x * (blockDim.x >> 5) + wave;
  if (n >= NNODES) return;
  const int b = batch[n];
  const float coef = gbuf[n] / gden[b];
  atomicAdd(&pool[(size_t)b * 64 + lane], coef * x[(size_t)n * 64 + lane]);
  atomicAdd(&pool[(size_t)b * 64 + lane + 32], coef * x[(size_t)n * 64 + lane + 32]);
}

// ---------------- graph-level head ----------------
__global__ void k_head(const float* __restrict__ x1, const float* __restrict__ x2,
                       const float* __restrict__ w0, const float* __restrict__ c0,
                       const float* __restrict__ w1, const float* __restrict__ c1,
                       const float* __restrict__ w2, const float* __restrict__ c2,
                       const float* __restrict__ w3, const float* __restrict__ c3,
                       float* __restrict__ out) {
  const int g = blockIdx.x;
  const int t = threadIdx.x;                   // 128 threads
  __shared__ float h[128], t1[128], t2[16], t3[16];
  h[t] = (t < 64) ? x1[(size_t)g * 64 + t] : x2[(size_t)g * 64 + (t - 64)];
  __syncthreads();
  float s = c0[t];
  for (int i = 0; i < 128; ++i) s += h[i] * w0[i * 128 + t];
  t1[t] = s > 0.f ? s : 0.f;
  __syncthreads();
  if (t < 16) {
    float s2 = c1[t];
    for (int i = 0; i < 128; ++i) s2 += t1[i] * w1[i * 16 + t];
    t2[t] = s2 > 0.f ? s2 : 0.f;
  }
  __syncthreads();
  if (t < 16) {
    float s3 = c2[t];
    for (int i = 0; i < 16; ++i) s3 += t2[i] * w2[i * 16 + t];
    t3[t] = s3;                                // no relu on lins[2]
  }
  __syncthreads();
  if (t == 0) {
    float s4 = c3[0];
    for (int i = 0; i < 16; ++i) s4 += t3[i] * w3[i];
    out[g] = s4;
  }
}

// ---------------- host orchestration ----------------
template<int CI, int CO>
static void run_mpl(const float* xin, float* agg, float* xnb,
                    const int* src, const int* dst,
                    void* const* d_in, int pb, char* ws, hipStream_t stream) {
  const float* W1 = (const float*)d_in[pb + 0];
  const float* B1 = (const float*)d_in[pb + 1];
  const float* W2 = (const float*)d_in[pb + 2];
  const float* B2 = (const float*)d_in[pb + 3];
  const float* W3 = (const float*)d_in[pb + 4];
  const float* B3 = (const float*)d_in[pb + 5];
  const float* W4 = (const float*)d_in[pb + 6];
  const float* B4 = (const float*)d_in[pb + 7];
  const float* W5 = (const float*)d_in[pb + 8];
  const float* B5 = (const float*)d_in[pb + 9];
  const float* W6 = (const float*)d_in[pb + 10];
  const float* B6 = (const float*)d_in[pb + 11];
  const float* W7 = (const float*)d_in[pb + 12];
  const float* B7 = (const float*)d_in[pb + 13];

  __bf16* w1t = (__bf16*)(ws + OFF_W1T);
  __bf16* w2t = (__bf16*)(ws + OFF_W2T);
  __bf16* w5t = (__bf16*)(ws + OFF_W5T);
  __bf16* w6t = (__bf16*)(ws + OFF_W6T);
  float* ew = (float*)(ws + OFF_EW);
  unsigned* segmax = (unsigned*)(ws + OFF_SEGMX);
  float* denom = (float*)(ws + OFF_DENOM);

  k_transpose_bf16<<<(2 * CI * CO + 255) / 256, 256, 0, stream>>>(w1t, W1, 2 * CI, CO);
  k_transpose_bf16<<<(CI * CO + 255) / 256, 256, 0, stream>>>(w2t, W2, CI, CO);
  k_transpose_bf16<<<(CI * 16 + 255) / 256, 256, 0, stream>>>(w5t, W5, CI, 16);
  k_transpose_bf16<<<(CO * 16 + 255) / 256, 256, 0, stream>>>(w6t, W6, CO, 16);

  hipMemsetAsync(segmax, 0, (size_t)NNODES * 4, stream);  // enc(0) <= enc(-inf..)
  hipMemsetAsync(denom, 0, (size_t)NNODES * 4, stream);
  hipMemsetAsync(agg, 0, (size_t)NNODES * CO * 4, stream);

  const int etiles = NEDGES / 16;                           // 30000
  k_edge_pass1<CI, CO><<<etiles / 4, 128, 0, stream>>>(
      xin, src, dst, w1t, B1, w5t, B5, w6t, B6, W7, B7, ew, segmax, etiles);
  k_edge_exp<<<(NEDGES + 255) / 256, 256, 0, stream>>>(ew, segmax, denom, src);
  k_edge_div<<<(NEDGES + 255) / 256, 256, 0, stream>>>(ew, denom, src);
  k_edge_pass2<CI, CO><<<etiles / 4, 128, 0, stream>>>(
      xin, src, dst, w1t, B1, ew, agg, etiles);

  const int ntiles = NNODES / 16;                           // 1875
  k_node_xn<CI, CO><<<(ntiles + 3) / 4, 128, 0, stream>>>(xin, w2t, B2, xnb, ntiles);
  k_combine<CO><<<NNODES / 8, 256, 0, stream>>>(xnb, agg, W3, B3, W4, B4, NNODES);
}

static void att_pool(const float* x, const float* gw, const float* gb,
                     const int* batch, float* pool, char* ws, hipStream_t stream) {
  float* gbuf = (float*)(ws + OFF_EW);
  unsigned* gmax = (unsigned*)(ws + OFF_SEGMX);
  float* gden = (float*)(ws + OFF_DENOM);
  hipMemsetAsync(gmax, 0, NGRAPHS * 4, stream);
  hipMemsetAsync(gden, 0, NGRAPHS * 4, stream);
  hipMemsetAsync(pool, 0, (size_t)NGRAPHS * 64 * 4, stream);
  k_pool_g<<<NNODES / 8, 256, 0, stream>>>(x, gw, gb, batch, gbuf, gmax);
  k_pool_e<<<(NNODES + 255) / 256, 256, 0, stream>>>(gbuf, gmax, gden, batch);
  k_pool_acc<<<NNODES / 8, 256, 0, stream>>>(x, gbuf, gden, batch, pool);
}

extern "C" void kernel_launch(void* const* d_in, const int* in_sizes, int n_in,
                              void* d_out, int out_size, void* d_ws, size_t ws_size,
                              hipStream_t stream) {
  (void)in_sizes; (void)n_in; (void)out_size; (void)ws_size;
  const float* x = (const float*)d_in[0];
  const int* ei = (const int*)d_in[1];
  const int* batch = (const int*)d_in[2];
  const int* src = ei;              // edge_index[0]
  const int* dst = ei + NEDGES;     // edge_index[1]
  char* ws = (char*)d_ws;
  float* B0 = (float*)(ws + OFF_BIG0);
  float* B1 = (float*)(ws + OFF_BIG1);
  float* B2 = (float*)(ws + OFF_BIG2);
  float* x1p = (float*)(ws + OFF_X1);
  float* x2p = (float*)(ws + OFF_X2);

  // param bases in flattened dict order: conv1=3, conv2=17, conv3=31, conv4=45,
  // lins=59..66, gate1=67, gate2=69
  run_mpl<64, 128>(x,  B0, B1, src, dst, d_in, 3,  ws, stream);  // h1 -> B0
  run_mpl<128, 64>(B0, B2, B1, src, dst, d_in, 17, ws, stream);  // h2 -> B2
  att_pool(B2, (const float*)d_in[67], (const float*)d_in[68], batch, x1p, ws, stream);
  run_mpl<64, 64>(B2, B0, B1, src, dst, d_in, 31, ws, stream);   // h3 -> B0
  run_mpl<64, 64>(B0, B2, B1, src, dst, d_in, 45, ws, stream);   // h4 -> B2
  att_pool(B2, (const float*)d_in[69], (const float*)d_in[70], batch, x2p, ws, stream);

  k_head<<<NGRAPHS, 128, 0, stream>>>(
      x1p, x2p,
      (const float*)d_in[59], (const float*)d_in[60],
      (const float*)d_in[61], (const float*)d_in[62],
      (const float*)d_in[63], (const float*)d_in[64],
      (const float*)d_in[65], (const float*)d_in[66],
      (float*)d_out);
}
